// Embeddings_1151051235393
// MI455X (gfx1250) — matrix operand order repaired and checked by reading, not compiled, not run on hardware
//
#include <hip/hip_runtime.h>
#include <hip/hip_bf16.h>

// ---------------------------------------------------------------------------
// Problem constants (from reference)
// ---------------------------------------------------------------------------
#define BATCH 64
#define C_IN  3
#define IMG   384
#define PATCH 16
#define HID   768
#define CQ    96
#define HW    24              // H = W = 24
#define NP    576             // 24*24
#define MDIM  (BATCH * NP)    // 36864 GEMM rows
#define KDIM  768             // im2col K and hidden K

typedef __attribute__((ext_vector_type(16))) __bf16 v16bf;
typedef __attribute__((ext_vector_type(8)))  float  v8f;

union Frag { uint4 u[2]; v16bf v; };

__device__ __forceinline__ unsigned short f2bf(float f) {
    unsigned u = __builtin_bit_cast(unsigned, f);
    unsigned r = u + 0x7FFFu + ((u >> 16) & 1u);   // round-to-nearest-even
    return (unsigned short)(r >> 16);
}
__device__ __forceinline__ float bf2f(unsigned short h) {
    return __builtin_bit_cast(float, (unsigned)h << 16);
}

// ---------------------------------------------------------------------------
// fp32 -> bf16 copy (weight conversion)
// ---------------------------------------------------------------------------
__global__ void cvt_bf16_kernel(const float* __restrict__ src,
                                unsigned short* __restrict__ dst, int n) {
    int i = blockIdx.x * blockDim.x + threadIdx.x;
    if (i < n) dst[i] = f2bf(src[i]);
}

// ---------------------------------------------------------------------------
// im2col: x (B,3,384,384) f32 -> Xp (36864, 768) bf16 ; k = (c, ph, pw)
// ---------------------------------------------------------------------------
__global__ void im2col_kernel(const float* __restrict__ x,
                              unsigned short* __restrict__ Xp) {
    int idx = blockIdx.x * blockDim.x + threadIdx.x;
    if (idx >= MDIM * KDIM) return;
    int m = idx / KDIM, k = idx % KDIM;
    int b = m / NP, hw = m % NP;
    int h = hw / HW, w = hw % HW;
    int c = k / (PATCH * PATCH), rem = k % (PATCH * PATCH);
    int ph = rem / PATCH, pw = rem % PATCH;
    size_t src = (((size_t)(b * C_IN + c) * IMG) + (h * PATCH + ph)) * IMG
               + (w * PATCH + pw);
    Xp[idx] = f2bf(x[src]);
}

// ---------------------------------------------------------------------------
// bf16 WMMA GEMM:  C[M,N] = A[M,K] * B[N,K]^T (+bias, +epilogue)
// Block = 256 thr = 8 waves arranged WR x WC; wave tile (MI*16) x (NI*16).
// Block tile: (WR*MI*16) x (WC*NI*16).
// MODE 0: patch embed -> out_bf16 = Y (Y=acc+bias), out_f32 = Y + pos_emb
// MODE 1: Q|K        -> out_f32[m,192] = acc + (n<96 ? bq[n] : bk[n-96])
// MODE 2: V          -> out_bf16[m,768] = acc + bv[n]
// ---------------------------------------------------------------------------
template <int MODE, int MI, int NI, int WR, int WC>
__global__ __launch_bounds__(256)
void gemm_bf16_kernel(const unsigned short* __restrict__ A,
                      const unsigned short* __restrict__ Bw,
                      int K,
                      const float* __restrict__ bias0,
                      const float* __restrict__ bias1,
                      const float* __restrict__ posemb,
                      float* __restrict__ out_f32,
                      unsigned short* __restrict__ out_bf16) {
    const int lane = threadIdx.x & 31;
    const int wave = threadIdx.x >> 5;
    const int wr = wave % WR, wc = wave / WR;
    const int m0 = blockIdx.y * (WR * MI * 16) + wr * MI * 16;
    const int n0 = blockIdx.x * (WC * NI * 16) + wc * NI * 16;
    const int lrow = lane & 15;
    const int sel  = lane >> 4;

    const unsigned short* Ar[MI];
    const unsigned short* Br[NI];
#pragma unroll
    for (int mi = 0; mi < MI; ++mi)
        Ar[mi] = A + (size_t)(m0 + 16 * mi + lrow) * K;
#pragma unroll
    for (int ni = 0; ni < NI; ++ni)
        Br[ni] = Bw + (size_t)(n0 + 16 * ni + lrow) * K;

    v8f acc[MI][NI];
#pragma unroll
    for (int mi = 0; mi < MI; ++mi)
#pragma unroll
        for (int ni = 0; ni < NI; ++ni) acc[mi][ni] = v8f{};

    for (int k0 = 0; k0 < K; k0 += 32) {
        // A fragments: lane holds K chunks [k0+8*sel,+8) and [k0+16+8*sel,+8)
        const int ao0 = k0 + 8 * sel, ao1 = k0 + 16 + 8 * sel;
        // B fragments: lane holds K chunks [k0+8*sel,+8) and [k0+16+8*sel,+8)
        const int bo = k0 + 8 * sel;
        Frag a[MI], b[NI];
#pragma unroll
        for (int mi = 0; mi < MI; ++mi) {
            a[mi].u[0] = *(const uint4*)(Ar[mi] + ao0);
            a[mi].u[1] = *(const uint4*)(Ar[mi] + ao1);
        }
#pragma unroll
        for (int ni = 0; ni < NI; ++ni) {
            b[ni].u[0] = *(const uint4*)(Br[ni] + bo);
            b[ni].u[1] = *(const uint4*)(Br[ni] + bo + 16);
        }
#pragma unroll
        for (int mi = 0; mi < MI; ++mi)
#pragma unroll
            for (int ni = 0; ni < NI; ++ni)
                acc[mi][ni] = __builtin_amdgcn_wmma_f32_16x16x32_bf16(
                    false, a[mi].v, false, b[ni].v, (short)0, acc[mi][ni],
                    false, false);
    }

#pragma unroll
    for (int mi = 0; mi < MI; ++mi)
#pragma unroll
    for (int ni = 0; ni < NI; ++ni) {
        const int n = n0 + 16 * ni + lrow;
#pragma unroll
        for (int r = 0; r < 8; ++r) {
            const int m = m0 + 16 * mi + 8 * sel + r;
            float cv = acc[mi][ni][r];
            if (MODE == 0) {
                float yv = cv + bias0[n];
                out_bf16[(size_t)m * HID + n] = f2bf(yv);
                int p = m % NP;
                out_f32[(size_t)m * HID + n] = yv + posemb[(size_t)p * HID + n];
            } else if (MODE == 1) {
                float bv = (n < CQ) ? bias0[n] : bias1[n - CQ];
                out_f32[(size_t)m * (2 * CQ) + n] = cv + bv;
            } else {
                out_bf16[(size_t)m * HID + n] = f2bf(cv + bias0[n]);
            }
        }
    }
}

// ---------------------------------------------------------------------------
// energy_H: per (b, column c): eH[r][j] = dot96(Q[b,r,c], K[b,j,c]); diag=-1e30
// att layout: att[(b*576 + r*24 + c)*48 + j]
// ---------------------------------------------------------------------------
__global__ __launch_bounds__(256)
void energy_h_kernel(const float* __restrict__ QK, float* __restrict__ att) {
    const int b = blockIdx.x / HW, c = blockIdx.x % HW;
    __shared__ float Qs[HW][CQ + 1];
    __shared__ float Ks[HW][CQ + 1];
    for (int t = threadIdx.x; t < HW * CQ; t += 256) {
        int r = t / CQ, ch = t % CQ;
        size_t base = ((size_t)(b * NP) + r * HW + c) * (2 * CQ);
        Qs[r][ch] = QK[base + ch];
        Ks[r][ch] = QK[base + CQ + ch];
    }
    __syncthreads();
    for (int t = threadIdx.x; t < HW * HW; t += 256) {
        int r = t / HW, j = t % HW;
        float e = -1e30f;
        if (j != r) {
            e = 0.f;
            for (int ch = 0; ch < CQ; ++ch) e += Qs[r][ch] * Ks[j][ch];
        }
        att[((size_t)(b * NP) + r * HW + c) * 48 + j] = e;
    }
}

// energy_W: per (b, row r): eW[i][j] = dot96(Q[b,r,i], K[b,r,j]); no mask
__global__ __launch_bounds__(256)
void energy_w_kernel(const float* __restrict__ QK, float* __restrict__ att) {
    const int b = blockIdx.x / HW, r = blockIdx.x % HW;
    __shared__ float Qs[HW][CQ + 1];
    __shared__ float Ks[HW][CQ + 1];
    for (int t = threadIdx.x; t < HW * CQ; t += 256) {
        int i = t / CQ, ch = t % CQ;
        size_t base = ((size_t)(b * NP) + r * HW + i) * (2 * CQ);
        Qs[i][ch] = QK[base + ch];
        Ks[i][ch] = QK[base + CQ + ch];
    }
    __syncthreads();
    for (int t = threadIdx.x; t < HW * HW; t += 256) {
        int i = t / HW, j = t % HW;
        float e = 0.f;
        for (int ch = 0; ch < CQ; ++ch) e += Qs[i][ch] * Ks[j][ch];
        att[((size_t)(b * NP) + r * HW + i) * 48 + 24 + j] = e;
    }
}

// softmax over the 48 energies at each spatial position
__global__ void softmax48_kernel(float* __restrict__ att) {
    int p = blockIdx.x * blockDim.x + threadIdx.x;
    if (p >= MDIM) return;
    float* row = att + (size_t)p * 48;
    float mx = -1e38f;
    for (int i = 0; i < 48; ++i) mx = fmaxf(mx, row[i]);
    float s = 0.f;
    float e[48];
    for (int i = 0; i < 48; ++i) { e[i] = expf(row[i] - mx); s += e[i]; }
    float inv = 1.f / s;
    for (int i = 0; i < 48; ++i) row[i] = e[i] * inv;
}

// ---------------------------------------------------------------------------
// out_H: per (b, column c): d_out[b, (i,c), o] += g * sum_j attH[i,j] V[b,(j,c),o]
// ---------------------------------------------------------------------------
__global__ __launch_bounds__(256)
void out_h_kernel(const float* __restrict__ att,
                  const unsigned short* __restrict__ Vbf,
                  const float* __restrict__ gamma,
                  float* __restrict__ out) {
    const int b = blockIdx.x / HW, c = blockIdx.x % HW;
    const float g = gamma[0];
    __shared__ float Ws[HW][HW + 1];
    __shared__ float Vs[HW][128];
    for (int t = threadIdx.x; t < HW * HW; t += 256) {
        int i = t / HW, j = t % HW;
        Ws[i][j] = att[((size_t)(b * NP) + i * HW + c) * 48 + j];
    }
    __syncthreads();
    for (int o0 = 0; o0 < HID; o0 += 128) {
        for (int t = threadIdx.x; t < HW * 128; t += 256) {
            int j = t >> 7, oc = t & 127;
            Vs[j][oc] = bf2f(Vbf[((size_t)(b * NP) + j * HW + c) * HID + o0 + oc]);
        }
        __syncthreads();
        for (int t = threadIdx.x; t < HW * 128; t += 256) {
            int i = t >> 7, oc = t & 127;
            float s = 0.f;
            for (int j = 0; j < HW; ++j) s += Ws[i][j] * Vs[j][oc];
            size_t oi = ((size_t)(b * NP) + i * HW + c) * HID + o0 + oc;
            out[oi] += g * s;
        }
        __syncthreads();
    }
}

// out_W: per (b, row r): d_out[b, (r,i), o] += g * sum_j attW[i,j] V[b,(r,j),o]
__global__ __launch_bounds__(256)
void out_w_kernel(const float* __restrict__ att,
                  const unsigned short* __restrict__ Vbf,
                  const float* __restrict__ gamma,
                  float* __restrict__ out) {
    const int b = blockIdx.x / HW, r = blockIdx.x % HW;
    const float g = gamma[0];
    __shared__ float Ws[HW][HW + 1];
    __shared__ float Vs[HW][128];
    for (int t = threadIdx.x; t < HW * HW; t += 256) {
        int i = t / HW, j = t % HW;
        Ws[i][j] = att[((size_t)(b * NP) + r * HW + i) * 48 + 24 + j];
    }
    __syncthreads();
    for (int o0 = 0; o0 < HID; o0 += 128) {
        for (int t = threadIdx.x; t < HW * 128; t += 256) {
            int j = t >> 7, oc = t & 127;
            Vs[j][oc] = bf2f(Vbf[((size_t)(b * NP) + r * HW + j) * HID + o0 + oc]);
        }
        __syncthreads();
        for (int t = threadIdx.x; t < HW * 128; t += 256) {
            int i = t >> 7, oc = t & 127;
            float s = 0.f;
            for (int j = 0; j < HW; ++j) s += Ws[i][j] * Vs[j][oc];
            size_t oi = ((size_t)(b * NP) + r * HW + i) * HID + o0 + oc;
            out[oi] += g * s;
        }
        __syncthreads();
    }
}

// ---------------------------------------------------------------------------
// Host-side orchestration
// ---------------------------------------------------------------------------
extern "C" void kernel_launch(void* const* d_in, const int* in_sizes, int n_in,
                              void* d_out, int out_size, void* d_ws, size_t ws_size,
                              hipStream_t stream) {
    const float* x       = (const float*)d_in[0];
    const float* patch_w = (const float*)d_in[1];
    const float* patch_b = (const float*)d_in[2];
    const float* wq      = (const float*)d_in[3];
    const float* bq      = (const float*)d_in[4];
    const float* wk      = (const float*)d_in[5];
    const float* bk      = (const float*)d_in[6];
    const float* wv      = (const float*)d_in[7];
    const float* bv      = (const float*)d_in[8];
    const float* gamma   = (const float*)d_in[9];
    const float* pos_emb = (const float*)d_in[10];
    float* out = (float*)d_out;

    // workspace carve-out (256B aligned)
    char* w = (char*)d_ws;
    auto alloc = [&](size_t bytes) -> void* {
        void* p = (void*)w;
        w += (bytes + 255) & ~(size_t)255;
        return p;
    };
    unsigned short* Xp  = (unsigned short*)alloc((size_t)MDIM * KDIM * 2);   // 56.6 MB
    unsigned short* Wp  = (unsigned short*)alloc((size_t)HID * KDIM * 2);    // 1.2 MB
    unsigned short* Wqk = (unsigned short*)alloc((size_t)2 * CQ * KDIM * 2); // 0.3 MB
    unsigned short* Wv  = (unsigned short*)alloc((size_t)HID * KDIM * 2);    // 1.2 MB
    unsigned short* Ybf = (unsigned short*)alloc((size_t)MDIM * HID * 2);    // 56.6 MB
    float*          QKb = (float*)alloc((size_t)MDIM * 2 * CQ * 4);          // 28.3 MB
    unsigned short* Vbf = (unsigned short*)alloc((size_t)MDIM * HID * 2);    // 56.6 MB
    float*          attb= (float*)alloc((size_t)MDIM * 48 * 4);              // 7.1 MB

    // 1) weight conversions to bf16
    {
        int n = HID * KDIM;
        cvt_bf16_kernel<<<(n + 255) / 256, 256, 0, stream>>>(patch_w, Wp, n);
        int nq = CQ * KDIM;
        cvt_bf16_kernel<<<(nq + 255) / 256, 256, 0, stream>>>(wq, Wqk, nq);
        cvt_bf16_kernel<<<(nq + 255) / 256, 256, 0, stream>>>(wk, Wqk + (size_t)CQ * KDIM, nq);
        cvt_bf16_kernel<<<(n + 255) / 256, 256, 0, stream>>>(wv, Wv, n);
    }

    // 2) im2col of x into bf16 A matrix
    {
        int total = MDIM * KDIM;
        im2col_kernel<<<(total + 255) / 256, 256, 0, stream>>>(x, Xp);
    }

    // 3) patch-embed GEMM: Ybf = Xp @ Wp^T + b ; d_out = Y + pos_emb
    //    waves 2(M)x4(N), wave tile 64x32 -> block tile 128x128
    gemm_bf16_kernel<0, 4, 2, 2, 4><<<dim3(HID / 128, MDIM / 128), 256, 0, stream>>>(
        Xp, Wp, KDIM, patch_b, nullptr, pos_emb, out, Ybf);

    // 4) Q|K GEMM (N = 192): waves 4(M)x2(N), wave tile 32x16 -> block 128x32
    gemm_bf16_kernel<1, 2, 1, 4, 2><<<dim3((2 * CQ) / 32, MDIM / 128), 256, 0, stream>>>(
        Ybf, Wqk, KDIM, bq, bk, nullptr, QKb, nullptr);

    // 5) V GEMM (N = 768): same geometry as patch GEMM
    gemm_bf16_kernel<2, 4, 2, 2, 4><<<dim3(HID / 128, MDIM / 128), 256, 0, stream>>>(
        Ybf, Wv, KDIM, bv, nullptr, nullptr, nullptr, Vbf);

    // 6) energies + softmax
    energy_h_kernel<<<BATCH * HW, 256, 0, stream>>>(QKb, attb);
    energy_w_kernel<<<BATCH * HW, 256, 0, stream>>>(QKb, attb);
    softmax48_kernel<<<MDIM / 256, 256, 0, stream>>>(attb);

    // 7) attention outputs accumulated into d_out (scaled by gamma)
    out_h_kernel<<<BATCH * HW, 256, 0, stream>>>(attb, Vbf, gamma, out);
    out_w_kernel<<<BATCH * HW, 256, 0, stream>>>(attb, Vbf, gamma, out);

    (void)in_sizes; (void)n_in; (void)out_size; (void)ws_size;
}
